// LatentAttention_283467841576
// MI455X (gfx1250) — matrix operand order repaired and checked by reading, not compiled, not run on hardware
//
#include <hip/hip_runtime.h>
#include <stdint.h>

#define DM    1024
#define NH    8
#define NKV   2
#define HD    128
#define DL    32
#define BB    2
#define TT    2048
#define MROWS (BB*TT)   /* 4096 */

typedef __attribute__((ext_vector_type(16))) __bf16 v16bf;
typedef __attribute__((ext_vector_type(8)))  __bf16 v8bf;
typedef __attribute__((ext_vector_type(8)))  float  v8f;

#define WMMA_BF16(a,b,c) \
  __builtin_amdgcn_wmma_f32_16x16x32_bf16(false,(a),false,(b),(short)0,(c),false,false)

static __device__ __forceinline__ __bf16 f2bf(float f){
  unsigned u = __builtin_bit_cast(unsigned, f);
  u += 0x7fffu + ((u >> 16) & 1u);                 // round-to-nearest-even
  unsigned short h = (unsigned short)(u >> 16);
  return __builtin_bit_cast(__bf16, h);
}

static __device__ __forceinline__ v16bf comb(v8bf lo, v8bf hi){
  v16bf r;
#pragma unroll
  for (int i=0;i<8;i++){ r[i]=lo[i]; r[i+8]=hi[i]; }
  return r;
}

// 16 contiguous bf16 (two 16B LDS loads)
static __device__ __forceinline__ v16bf ld16(const __bf16* p){
  return comb(*(const v8bf*)p, *(const v8bf*)(p+8));
}

// A-fragment (16x32 bf16). contiguous-k layout used here: row = lane&15,
// K = e + 16*(lane>>4)  => one contiguous 16-elem run.
static __device__ __forceinline__ v16bf load_afrag(const __bf16* rowptr, int kbase, int half){
  return comb(*(const v8bf*)(rowptr + kbase + 16*half),
              *(const v8bf*)(rowptr + kbase + 16*half + 8));
}

// CDNA5 async global->LDS copy, 16 bytes per lane. Tracked by ASYNCcnt.
// Generic LDS pointer's low 32 bits are the LDS byte address (ISA 10.2:
// flat LDS addresses truncate to addr[31:0]).
static __device__ __forceinline__ void async_cp16(void* lds, const void* g){
  asm volatile("global_load_async_to_lds_b128 %0, %1, off"
               :: "v"((unsigned)(size_t)lds),
                  "v"((unsigned long long)(size_t)g)
               : "memory");
}
static __device__ __forceinline__ void wait_async0(){
  asm volatile("s_wait_asynccnt 0x0" ::: "memory");
}

// Transposed pair-pack: given dwords a={a0,a1} b={b0,b1} (bf16 pairs from two
// consecutive rows), emit {b0,a0} and {b1,a1} via single v_perm_b32 each.
static __device__ __forceinline__ unsigned perm_lo(unsigned hiw, unsigned low){
  return __builtin_amdgcn_perm(hiw, low, 0x05040100u);  // {hi.l, lo.l}
}
static __device__ __forceinline__ unsigned perm_hi(unsigned hiw, unsigned low){
  return __builtin_amdgcn_perm(hiw, low, 0x07060302u);  // {hi.h, lo.h}
}

__global__ void cvt_f32_bf16(const float* __restrict__ in, __bf16* __restrict__ out, int n){
  int i = blockIdx.x * blockDim.x + threadIdx.x;
  if (i < n) out[i] = f2bf(in[i]);
}

// ---------------------------------------------------------------------------
// Generic bf16 WMMA GEMM: C[M,N] = A[M,K] * B[K,N].
// Block tile 128x64, K-step 32, 8 waves (wave w owns rows 16w..16w+15).
// Double-buffered LDS, single barrier per k-step: A tile via async-to-LDS,
// B tile staged transposed ([col][k]) via v_perm pair-packed b32 stores so
// B-fragments are contiguous LDS reads (B layout used here: N=lane%16, K=e+16*(lane/16)).
// ---------------------------------------------------------------------------
__global__ __launch_bounds__(256)
void gemm_bf16_wmma(const __bf16* __restrict__ A, int lda,
                    const __bf16* __restrict__ Bm, int ldb,
                    void* __restrict__ C, int ldc,
                    int K, int storef32)
{
  __shared__ __bf16 As[2][128*32];   // 2 x 8 KB
  __shared__ __bf16 Bt[2][64*32];    // 2 x 4 KB, transposed [col][k]
  const int tid  = threadIdx.x;
  const int wave = tid >> 5, lane = tid & 31;
  const int half = lane >> 4, cl = lane & 15;
  const int m0 = blockIdx.x * 128, n0 = blockIdx.y * 64;

  auto stage = [&](int k0, int buf){
    { // A tile 128x32: async global->LDS, 32B per thread
      int row = tid >> 1, seg = (tid & 1) * 16;
      const __bf16* src = A + (size_t)(m0 + row) * lda + k0 + seg;
      async_cp16(&As[buf][row*32 + seg], src);
      async_cp16(&As[buf][row*32 + seg + 8], src + 8);
      if (k0 + 32 < K) __builtin_prefetch(src + 32, 0, 3);  // global_prefetch_b8
    }
    if (tid < 128) { // B tile 32x64 -> transposed, v_perm pair-packed b32 stores
      int kp = tid >> 3, c0 = (tid & 7) * 8;     // k-pair 0..15, col chunk of 8
      const __bf16* s0p = Bm + (size_t)(k0 + 2*kp) * ldb + n0 + c0;
      uint4 r0 = *(const uint4*)s0p;
      uint4 r1 = *(const uint4*)(s0p + ldb);
      const unsigned* a = (const unsigned*)&r0;
      const unsigned* b = (const unsigned*)&r1;
      unsigned* dst = (unsigned*)&Bt[buf][c0*32 + 2*kp];  // stride 32 bf16 = 16 dwords
#pragma unroll
      for (int wd=0; wd<4; ++wd){
        dst[(2*wd  )*16] = perm_lo(b[wd], a[wd]);
        dst[(2*wd+1)*16] = perm_hi(b[wd], a[wd]);
      }
    }
  };

  v8f acc[4];
#pragma unroll
  for (int n=0;n<4;n++)
#pragma unroll
    for (int r=0;r<8;r++) acc[n][r] = 0.0f;

  const int nk = K >> 5;
  stage(0, 0);
  for (int i = 0; i < nk; ++i) {
    wait_async0();
    __syncthreads();
    if (i + 1 < nk) stage((i + 1) << 5, (i + 1) & 1);
    const __bf16* as = As[i & 1];
    const __bf16* bt = Bt[i & 1];
    v16bf a = load_afrag(&as[(wave*16 + cl)*32], 0, half);
#pragma unroll
    for (int n=0;n<4;n++){
      v16bf b = ld16(&bt[(n*16 + cl)*32 + 16*half]);
      acc[n] = WMMA_BF16(a, b, acc[n]);
    }
  }

#pragma unroll
  for (int n=0;n<4;n++)
#pragma unroll
    for (int r=0;r<8;r++){
      int row = m0 + wave*16 + r + 8*half;   // C-layout: M = r + 8*(lane/16)
      int col = n0 + n*16 + cl;              // N = lane%16
      if (storef32) ((float*)C)[(size_t)row*ldc + col] = acc[n][r];
      else          ((__bf16*)C)[(size_t)row*ldc + col] = f2bf(acc[n][r]);
    }
}

// ---------------------------------------------------------------------------
// Causal flash attention, GQA (q:[B,T,8,128] bf16, k/v:[B,T,2,128] bf16).
// Block: 128 query rows, 8 waves x 16 rows. Key blocks of 32, double-buffered
// K/V staging (K async row-major; V transposed via v_perm pair-packing),
// single barrier per key block.
// ---------------------------------------------------------------------------
__global__ __launch_bounds__(256)
void mla_attn_wmma(const __bf16* __restrict__ qb, const __bf16* __restrict__ kb,
                   const __bf16* __restrict__ vb, __bf16* __restrict__ yb)
{
  __shared__ __bf16 Ks[2][32*HD];    // 2 x 8 KB, [key][dim]
  __shared__ __bf16 Vt[2][HD*32];    // 2 x 8 KB, transposed [dim][key]
  __shared__ __bf16 Ps[8*16*32];     // 8 KB, per-wave P re-layout buffers

  const int tid = threadIdx.x, wave = tid >> 5, lane = tid & 31;
  const int half = lane >> 4, cl = lane & 15;
  const int qtile = blockIdx.x;
  const int b = blockIdx.y >> 3, h = blockIdx.y & 7, hkv = h >> 2;
  const int qbase = qtile * 128;
  const int qw = qbase + wave * 16;
  const float scale = 0.088388347648318447f;  // 1/sqrt(128)
  const int nkb = qtile * 4 + 4;              // keys 0 .. qbase+127

  auto stageKV = [&](int kbi, int buf){
    const int kstart = kbi * 32;
    { // K tile 32x128: async global->LDS, 32B per thread
      int key = tid >> 3, seg = (tid & 7) * 16;
      const __bf16* ksrc = kb + ((size_t)(b*TT + kstart + key) * NKV + hkv) * HD + seg;
      async_cp16(&Ks[buf][key*HD + seg], ksrc);
      async_cp16(&Ks[buf][key*HD + seg + 8], ksrc + 8);
      if (kbi + 1 < nkb) __builtin_prefetch(ksrc + 32*NKV*HD, 0, 3);
    }
    { // V tile -> transposed [dim][key], v_perm key-pair packed b32 stores
      int p = tid >> 4;            // key pair 0..15
      int vseg = (tid & 15) * 8;   // dim chunk of 8
      const __bf16* v0p = vb + ((size_t)(b*TT + kstart + 2*p) * NKV + hkv) * HD + vseg;
      uint4 r0 = *(const uint4*)v0p;
      uint4 r1 = *(const uint4*)(v0p + NKV*HD);
      const unsigned* a = (const unsigned*)&r0;
      const unsigned* c = (const unsigned*)&r1;
      unsigned* dst = (unsigned*)&Vt[buf][vseg*32 + 2*p];  // stride 32 bf16 = 16 dwords
#pragma unroll
      for (int wd=0; wd<4; ++wd){
        dst[(2*wd  )*16] = perm_lo(c[wd], a[wd]);
        dst[(2*wd+1)*16] = perm_hi(c[wd], a[wd]);
      }
    }
  };

  // Q fragments straight from global (this lane's row, 4 k-chunks of 32)
  const __bf16* qrow = qb + ((size_t)(b*TT + qw + cl) * NH + h) * HD;
  v16bf qa[4];
#pragma unroll
  for (int c=0;c<4;c++) qa[c] = load_afrag(qrow, 32*c, half);

  float mrow[8], lrow[8];
#pragma unroll
  for (int r=0;r<8;r++){ mrow[r] = -1e30f; lrow[r] = 0.0f; }
  v8f oacc[8];
#pragma unroll
  for (int c=0;c<8;c++)
#pragma unroll
    for (int r=0;r<8;r++) oacc[c][r] = 0.0f;

  __bf16* pw = &Ps[wave * (16*32)];

  stageKV(0, 0);
  for (int kbi = 0; kbi < nkb; ++kbi) {
    const int kstart = kbi * 32;
    wait_async0();
    __syncthreads();
    if (kbi + 1 < nkb) stageKV(kbi + 1, (kbi + 1) & 1);
    const __bf16* ks = Ks[kbi & 1];
    const __bf16* vt = Vt[kbi & 1];

    if (kstart <= qw + 15) {   // wave-uniform causal skip
      v8f s0, s1;
#pragma unroll
      for (int r=0;r<8;r++){ s0[r]=0.0f; s1[r]=0.0f; }
#pragma unroll
      for (int c=0;c<4;c++){
        // B frag for S: N=key (lane%16), K=dim chunk (contiguous in Ks row)
        v16bf b0 = ld16(&ks[cl*HD       + 32*c + 16*half]);
        v16bf b1 = ld16(&ks[(16+cl)*HD  + 32*c + 16*half]);
        s0 = WMMA_BF16(qa[c], b0, s0);
        s1 = WMMA_BF16(qa[c], b1, s1);
      }
      float rs[8];
#pragma unroll
      for (int r=0;r<8;r++){
        int qr = qw + r + 8*half;
        float x0 = (kstart + cl      <= qr) ? s0[r]*scale : -1e30f;
        float x1 = (kstart + 16 + cl <= qr) ? s1[r]*scale : -1e30f;
        float mx = fmaxf(x0, x1);
#pragma unroll
        for (int d=1; d<16; d<<=1) mx = fmaxf(mx, __shfl_xor(mx, d, 32));
        float mnew = fmaxf(mrow[r], mx);
        float sc = __expf(mrow[r] - mnew);
        float e0 = __expf(x0 - mnew), e1 = __expf(x1 - mnew);
        float ls = e0 + e1;
#pragma unroll
        for (int d=1; d<16; d<<=1) ls += __shfl_xor(ls, d, 32);
        lrow[r] = lrow[r]*sc + ls;
        mrow[r] = mnew;
        rs[r] = sc;
        // scatter P into per-wave LDS buffer in plain [row][key] layout
        pw[(r + 8*half)*32 + cl]      = f2bf(e0);
        pw[(r + 8*half)*32 + 16 + cl] = f2bf(e1);
      }
#pragma unroll
      for (int c=0;c<8;c++)
#pragma unroll
        for (int r=0;r<8;r++) oacc[c][r] *= rs[r];
      asm volatile("" ::: "memory");   // LDS in-order per wave; stop compiler reordering
      v16bf pa = load_afrag(&pw[cl*32], 0, half);
#pragma unroll
      for (int c=0;c<8;c++){
        // B frag for PV: N=out dim, K=key -> contiguous in transposed Vt
        v16bf vf = ld16(&vt[(c*16 + cl)*32 + 16*half]);
        oacc[c] = WMMA_BF16(pa, vf, oacc[c]);
      }
    }
  }

  float invl[8];
#pragma unroll
  for (int r=0;r<8;r++) invl[r] = 1.0f / lrow[r];
#pragma unroll
  for (int c=0;c<8;c++)
#pragma unroll
    for (int r=0;r<8;r++){
      int t = qw + r + 8*half;
      yb[((size_t)(b*TT + t) * NH + h) * HD + c*16 + cl] = f2bf(oacc[c][r] * invl[r]);
    }
}

// ---------------------------------------------------------------------------
extern "C" void kernel_launch(void* const* d_in, const int* in_sizes, int n_in,
                              void* d_out, int out_size, void* d_ws, size_t ws_size,
                              hipStream_t stream)
{
  const float* x   = (const float*)d_in[0];
  const float* wq  = (const float*)d_in[1];
  const float* wkd = (const float*)d_in[2];
  const float* wvd = (const float*)d_in[3];
  const float* wku = (const float*)d_in[4];
  const float* wvu = (const float*)d_in[5];
  const float* wo  = (const float*)d_in[6];

  char* w = (char*)d_ws;
  auto take = [&](size_t elems)->__bf16*{
    __bf16* p = (__bf16*)w;
    size_t bytes = (elems * sizeof(unsigned short) + 255) & ~(size_t)255;
    w += bytes; return p;
  };
  __bf16* xb   = take((size_t)MROWS*DM);
  __bf16* wqb  = take((size_t)DM*DM);
  __bf16* wkdb = take((size_t)DM*NKV*DL);
  __bf16* wvdb = take((size_t)DM*NKV*DL);
  __bf16* wkub = take((size_t)DL*HD);
  __bf16* wvub = take((size_t)DL*HD);
  __bf16* wob  = take((size_t)DM*DM);
  __bf16* qbuf = take((size_t)MROWS*DM);        // [B,T,8,128]
  __bf16* klat = take((size_t)MROWS*NKV*DL);    // [B,T,2,32]
  __bf16* vlat = take((size_t)MROWS*NKV*DL);
  __bf16* kbuf = take((size_t)MROWS*NKV*HD);    // [B,T,2,128]
  __bf16* vbuf = take((size_t)MROWS*NKV*HD);
  __bf16* ybuf = take((size_t)MROWS*DM);        // [B,T,8,128]

  auto cvt = [&](const float* src, __bf16* dst, int n){
    cvt_f32_bf16<<<(n + 255) / 256, 256, 0, stream>>>(src, dst, n);
  };
  cvt(x,   xb,   MROWS*DM);
  cvt(wq,  wqb,  DM*DM);
  cvt(wkd, wkdb, DM*NKV*DL);
  cvt(wvd, wvdb, DM*NKV*DL);
  cvt(wku, wkub, DL*HD);
  cvt(wvu, wvub, DL*HD);
  cvt(wo,  wob,  DM*DM);

  // q = x @ wq                               [4096,1024]x[1024,1024]
  gemm_bf16_wmma<<<dim3(MROWS/128, DM/64), 256, 0, stream>>>(xb, DM, wqb, DM, qbuf, DM, DM, 0);
  // k_lat / v_lat = x @ w*_down              [4096,1024]x[1024,64]
  gemm_bf16_wmma<<<dim3(MROWS/128, (NKV*DL)/64), 256, 0, stream>>>(xb, DM, wkdb, NKV*DL, klat, NKV*DL, DM, 0);
  gemm_bf16_wmma<<<dim3(MROWS/128, (NKV*DL)/64), 256, 0, stream>>>(xb, DM, wvdb, NKV*DL, vlat, NKV*DL, DM, 0);
  // latent up-projection per kv head         [4096,32]x[32,128] (strided A/C)
  for (int hh = 0; hh < NKV; ++hh) {
    gemm_bf16_wmma<<<dim3(MROWS/128, HD/64), 256, 0, stream>>>(klat + hh*DL, NKV*DL, wkub, HD, kbuf + hh*HD, NKV*HD, DL, 0);
    gemm_bf16_wmma<<<dim3(MROWS/128, HD/64), 256, 0, stream>>>(vlat + hh*DL, NKV*DL, wvub, HD, vbuf + hh*HD, NKV*HD, DL, 0);
  }
  // causal GQA attention -> ybuf bf16
  mla_attn_wmma<<<dim3(TT/128, BB*NH), 256, 0, stream>>>(qbuf, kbuf, vbuf, ybuf);
  // out = y @ wo -> f32 d_out                [4096,1024]x[1024,1024]
  gemm_bf16_wmma<<<dim3(MROWS/128, DM/64), 256, 0, stream>>>(ybuf, DM, wob, DM, d_out, DM, DM, 1);

  (void)in_sizes; (void)n_in; (void)out_size; (void)ws_size;
}
